// MultiFeatureRNN_38439957299381
// MI455X (gfx1250) — matrix operand order repaired and checked by reading, not compiled, not run on hardware
//
#include <hip/hip_runtime.h>

// ---------------------------------------------------------------------------
// Fused 3-layer Elman RNN for MI455X (gfx1250, wave32, WMMA bf16 -> f32 acc)
//   h_l,t = tanh(bias_l + h_{l-1},t @ W_ih_l^T + h_l,{t-1} @ W_hh_l^T)
// One wave owns 16 batch rows; all 3 hidden states live in LDS (bf16, 16x64
// padded). Weights are pre-packed into LDS in WMMA B-fragment layout.
// tanh uses gfx1250's hardware V_TANH_F32 (branch-free fallback otherwise).
// ---------------------------------------------------------------------------

typedef __attribute__((ext_vector_type(16))) __bf16 v16bf;
typedef __attribute__((ext_vector_type(8)))  float  v8f;

#define B_   4096
#define T_   256
#define H_   50
#define HP   64        // padded hidden
#define NT   4         // N tiles of 16
#define WAVES 2        // waves per workgroup
#define NWGS (B_ / (WAVES * 16))   // 128

__device__ __forceinline__ unsigned short f2bf(float f) {
    unsigned int u = __float_as_uint(f);
    u += 0x7FFFu + ((u >> 16) & 1u);          // round-to-nearest-even
    return (unsigned short)(u >> 16);
}
__device__ __forceinline__ float bf2f(unsigned short h) {
    return __uint_as_float(((unsigned int)h) << 16);
}

// Branch-free tanh. Prefer gfx1250 hardware v_tanh_f32.
__device__ __forceinline__ float fast_tanh(float x) {
#if __has_builtin(__builtin_amdgcn_tanhf)
    return __builtin_amdgcn_tanhf(x);
#else
    // tanh(x) = 1 - 2 / (exp2(2*log2(e)*x) + 1)
    // x->+inf: t=inf -> 1 ; x->-inf: t=0 -> -1 ; monotone, no EXEC branches.
#if __has_builtin(__builtin_amdgcn_exp2f)
    const float t = __builtin_amdgcn_exp2f(x * 2.8853900817779268f);
#else
    const float t = exp2f(x * 2.8853900817779268f);
#endif
#if __has_builtin(__builtin_amdgcn_rcpf)
    return 1.0f - 2.0f * __builtin_amdgcn_rcpf(t + 1.0f);
#else
    return 1.0f - 2.0f / (t + 1.0f);
#endif
#endif
}

__global__ void __launch_bounds__(WAVES * 32)
rnn3_fused_wmma(const float* __restrict__ x,
                const float* __restrict__ W_ih1, const float* __restrict__ W_hh1,
                const float* __restrict__ b_ih1, const float* __restrict__ b_hh1,
                const float* __restrict__ W_ih2, const float* __restrict__ W_hh2,
                const float* __restrict__ b_ih2, const float* __restrict__ b_hh2,
                const float* __restrict__ W_ih3, const float* __restrict__ W_hh3,
                const float* __restrict__ b_ih3, const float* __restrict__ b_hh3,
                const float* __restrict__ fc_w, const float* __restrict__ fc_b,
                float* __restrict__ out)
{
    // 5 matrices x 8 fragments x 32 lanes x 8 dwords (bf16 pairs) = 40 KB
    __shared__ __align__(32) unsigned int  s_w[5 * 8 * 32 * 8];
    // per-wave h staging: 3 layers x 16 rows x 64 cols bf16 = 6 KB/wave
    __shared__ __align__(16) unsigned short s_h[WAVES * 3 * 16 * HP];

    const int tid  = threadIdx.x;
    const int lane = tid & 31;
    const int wv   = tid >> 5;

    // ---- pack weights into LDS as WMMA B-fragments (B[k][n] = W[n][k]) ----
    // B 32x16 bf16 as packed here (contiguous k, same order as loadA): lane = (k/16)*16 + n%16 ; dword = (k%16)/2 ; half = k%2
    const float* mats[5] = { W_hh1, W_ih2, W_hh2, W_ih3, W_hh3 };
    for (int idx = tid; idx < 5 * 2048; idx += WAVES * 32) {
        const int mat  = idx >> 11;
        const int frag = (idx >> 8) & 7;          // nt*2 + kc
        const int ln   = (idx >> 3) & 31;
        const int dw   = idx & 7;
        const int nt = frag >> 1, kc = frag & 1;
        const int n  = nt * 16 + (ln & 15);
        const int k0 = kc * 32 + ((ln >> 4) << 4) + dw * 2;
        const float* W = mats[mat];
        const float v0 = (n < H_ && k0     < H_) ? W[n * H_ + k0]     : 0.f;
        const float v1 = (n < H_ && k0 + 1 < H_) ? W[n * H_ + k0 + 1] : 0.f;
        s_w[idx] = (unsigned int)f2bf(v0) | ((unsigned int)f2bf(v1) << 16);
    }
    // ---- zero hidden-state staging (h_{-1} = 0) ----
    {
        unsigned int* p = (unsigned int*)s_h;
        for (int i = tid; i < WAVES * 3 * 16 * HP / 2; i += WAVES * 32) p[i] = 0u;
    }
    __syncthreads();

    // ---- per-lane constants ----
    float bias1[NT], bias2[NT], bias3[NT], wi0[NT], wi1[NT];
#pragma unroll
    for (int nt = 0; nt < NT; ++nt) {
        const int  n = nt * 16 + (lane & 15);
        const bool v = n < H_;
        bias1[nt] = v ? (b_ih1[n] + b_hh1[n]) : 0.f;
        bias2[nt] = v ? (b_ih2[n] + b_hh2[n]) : 0.f;
        bias3[nt] = v ? (b_ih3[n] + b_hh3[n]) : 0.f;
        wi0[nt]   = v ? W_ih1[n * 2 + 0] : 0.f;
        wi1[nt]   = v ? W_ih1[n * 2 + 1] : 0.f;
    }

    const int bbase = (blockIdx.x * WAVES + wv) * 16;
    const int mrow  = lane & 15;             // A-fragment: this lane's M row
    const int khalf = (lane >> 4) << 4;      // A-fragment K sub-offset (0 / 16)
    const int rbase = (lane >> 4) << 3;      // C layout: M = rbase + r

    unsigned short* st0 = s_h + (wv * 3 + 0) * 16 * HP;
    unsigned short* st1 = s_h + (wv * 3 + 1) * 16 * HP;
    unsigned short* st2 = s_h + (wv * 3 + 2) * 16 * HP;

    // A-fragment load: two aligned b128 from row-major bf16 staging
    auto loadA = [&](const unsigned short* s, int c) {
        union { v16bf v; uint4 q[2]; } a;
        const unsigned short* p = s + mrow * HP + c * 32 + khalf;
        a.q[0] = *(const uint4*)(p);
        a.q[1] = *(const uint4*)(p + 8);
        return a.v;
    };
    // B-fragment load: lane-contiguous 32 bytes
    auto loadB = [&](int mat, int nt, int kc) {
        const unsigned int* p =
            s_w + ((((mat << 3) + (nt << 1) + kc) << 5) + lane) * 8;
        union { v16bf v; uint4 q[2]; } b;
        b.q[0] = *(const uint4*)(p);
        b.q[1] = *(const uint4*)(p + 4);
        return b.v;
    };
    // tanh + bf16-pack a C tile back into staging (row-major)
    auto storeC = [&](unsigned short* s, int nt, const v8f& c) {
#pragma unroll
        for (int r = 0; r < 8; ++r)
            s[(rbase + r) * HP + nt * 16 + (lane & 15)] = f2bf(fast_tanh(c[r]));
    };

    const float2* x2 = (const float2*)x;     // x[b][t][0..1]

    for (int t = 0; t < T_; ++t) {
        // ================= layer 1 =================
        const v16bf a1_0 = loadA(st0, 0);    // h1_{t-1}
        const v16bf a1_1 = loadA(st0, 1);
        float2 xr[8];
#pragma unroll
        for (int r = 0; r < 8; ++r)
            xr[r] = x2[(bbase + rbase + r) * T_ + t];
#pragma unroll
        for (int nt = 0; nt < NT; ++nt) {
            v8f c;
#pragma unroll
            for (int r = 0; r < 8; ++r)
                c[r] = bias1[nt] + xr[r].x * wi0[nt] + xr[r].y * wi1[nt];
            c = __builtin_amdgcn_wmma_f32_16x16x32_bf16(false, a1_0, false, loadB(0, nt, 0), (short)0, c, false, false);
            c = __builtin_amdgcn_wmma_f32_16x16x32_bf16(false, a1_1, false, loadB(0, nt, 1), (short)0, c, false, false);
            storeC(st0, nt, c);              // st0 <- h1_t
        }
        // ================= layer 2 =================
        const v16bf aL0 = loadA(st0, 0);     // h1_t (just written; LDS in-order)
        const v16bf aL1 = loadA(st0, 1);
        const v16bf a20 = loadA(st1, 0);     // h2_{t-1}
        const v16bf a21 = loadA(st1, 1);
#pragma unroll
        for (int nt = 0; nt < NT; ++nt) {
            v8f c;
#pragma unroll
            for (int r = 0; r < 8; ++r) c[r] = bias2[nt];
            c = __builtin_amdgcn_wmma_f32_16x16x32_bf16(false, aL0, false, loadB(1, nt, 0), (short)0, c, false, false);
            c = __builtin_amdgcn_wmma_f32_16x16x32_bf16(false, aL1, false, loadB(1, nt, 1), (short)0, c, false, false);
            c = __builtin_amdgcn_wmma_f32_16x16x32_bf16(false, a20, false, loadB(2, nt, 0), (short)0, c, false, false);
            c = __builtin_amdgcn_wmma_f32_16x16x32_bf16(false, a21, false, loadB(2, nt, 1), (short)0, c, false, false);
            storeC(st1, nt, c);              // st1 <- h2_t
        }
        // ================= layer 3 =================
        const v16bf aM0 = loadA(st1, 0);     // h2_t
        const v16bf aM1 = loadA(st1, 1);
        const v16bf a30 = loadA(st2, 0);     // h3_{t-1}
        const v16bf a31 = loadA(st2, 1);
#pragma unroll
        for (int nt = 0; nt < NT; ++nt) {
            v8f c;
#pragma unroll
            for (int r = 0; r < 8; ++r) c[r] = bias3[nt];
            c = __builtin_amdgcn_wmma_f32_16x16x32_bf16(false, aM0, false, loadB(3, nt, 0), (short)0, c, false, false);
            c = __builtin_amdgcn_wmma_f32_16x16x32_bf16(false, aM1, false, loadB(3, nt, 1), (short)0, c, false, false);
            c = __builtin_amdgcn_wmma_f32_16x16x32_bf16(false, a30, false, loadB(4, nt, 0), (short)0, c, false, false);
            c = __builtin_amdgcn_wmma_f32_16x16x32_bf16(false, a31, false, loadB(4, nt, 1), (short)0, c, false, false);
            storeC(st2, nt, c);              // st2 <- h3_t
        }
    }

    // ---- final FC: out[b][o] = h3_{T-1}[b] . fc_w[o] + fc_b[o] ----
    // 32 lanes -> 16 rows x 2 outputs, one result per lane
    {
        const int o = lane >> 4;             // 0..1
        const int m = lane & 15;
        float acc = fc_b[o];
        const unsigned short* h3 = st2 + m * HP;
        for (int h = 0; h < H_; ++h)
            acc += bf2f(h3[h]) * fc_w[o * H_ + h];
        out[(bbase + m) * 2 + o] = acc;
    }
}

extern "C" void kernel_launch(void* const* d_in, const int* in_sizes, int n_in,
                              void* d_out, int out_size, void* d_ws, size_t ws_size,
                              hipStream_t stream) {
    (void)in_sizes; (void)n_in; (void)d_ws; (void)ws_size; (void)out_size;
    const float* x     = (const float*)d_in[0];
    const float* W_ih1 = (const float*)d_in[1];
    const float* W_hh1 = (const float*)d_in[2];
    const float* b_ih1 = (const float*)d_in[3];
    const float* b_hh1 = (const float*)d_in[4];
    const float* W_ih2 = (const float*)d_in[5];
    const float* W_hh2 = (const float*)d_in[6];
    const float* b_ih2 = (const float*)d_in[7];
    const float* b_hh2 = (const float*)d_in[8];
    const float* W_ih3 = (const float*)d_in[9];
    const float* W_hh3 = (const float*)d_in[10];
    const float* b_ih3 = (const float*)d_in[11];
    const float* b_hh3 = (const float*)d_in[12];
    const float* fc_w  = (const float*)d_in[13];
    const float* fc_b  = (const float*)d_in[14];

    rnn3_fused_wmma<<<NWGS, WAVES * 32, 0, stream>>>(
        x, W_ih1, W_hh1, b_ih1, b_hh1,
        W_ih2, W_hh2, b_ih2, b_hh2,
        W_ih3, W_hh3, b_ih3, b_hh3,
        fc_w, fc_b, (float*)d_out);
}
